// PNAbstractionSet_34943853920280
// MI455X (gfx1250) — matrix operand order repaired and checked by reading, not compiled, not run on hardware
//
#include <hip/hip_runtime.h>
#include <hip/hip_bf16.h>

typedef __attribute__((ext_vector_type(16))) _Float16 v16h;
typedef __attribute__((ext_vector_type(8)))  float    v8f;

#define NB_B    8
#define NB_N    4096
#define NB_S    1024      // K_SAMPLES
#define NB_K    64        // K_NEIGH
#define R2      0.04f     // RADIUS^2
#define OUT_CH  131       // 3 + 128

// ---------------------------------------------------------------------------
// Kernel 1: farthest point sampling. One block per batch. Points staged in
// LDS (48KB << 320KB WGP LDS); per-thread min-dist in registers; argmax via
// wave32 shfl reduction + 8-entry LDS cross-wave reduce. Writes query coords
// to ws (for ball query) and directly into out[:, :, 0:3].
// ---------------------------------------------------------------------------
__global__ __launch_bounds__(256) void fps_kernel(const float* __restrict__ pts,
                                                  float* __restrict__ selected,
                                                  float* __restrict__ out) {
    const int b   = blockIdx.x;
    const int tid = threadIdx.x;
    const int wid = tid >> 5, lane = tid & 31;

    __shared__ float sx[NB_N], sy[NB_N], sz[NB_N];
    __shared__ float rv[8];
    __shared__ int   ri[8];
    __shared__ int   lastIdx;

    const float* P = pts + (size_t)b * NB_N * 3;
    for (int i = tid; i < NB_N; i += 256) {
        sx[i] = P[i * 3 + 0];
        sy[i] = P[i * 3 + 1];
        sz[i] = P[i * 3 + 2];
    }
    if (tid == 0) lastIdx = 0;
    __syncthreads();

    float mind[16];
#pragma unroll
    for (int j = 0; j < 16; ++j) mind[j] = 1e10f;   // reference "big"

    int cur = 0;
    for (int s = 0; s < NB_S; ++s) {
        const float cx = sx[cur], cy = sy[cur], cz = sz[cur];
        if (tid == 0) {
            float* sp = selected + ((size_t)b * NB_S + s) * 3;
            sp[0] = cx; sp[1] = cy; sp[2] = cz;
            float* orow = out + ((size_t)b * NB_S + s) * OUT_CH;
            orow[0] = cx; orow[1] = cy; orow[2] = cz;
        }
        if (s == NB_S - 1) break;

        float bestv = -1.0f;
        int   besti = 0;
#pragma unroll
        for (int j = 0; j < 16; ++j) {
            const int i = tid + j * 256;
            const float dx = sx[i] - cx, dy = sy[i] - cy, dz = sz[i] - cz;
            const float d = dx * dx + dy * dy + dz * dz;
            mind[j] = fminf(mind[j], d);
            if (mind[j] > bestv) { bestv = mind[j]; besti = i; }
        }
        // wave32 argmax (tie -> lowest index, matching jnp.argmax-first)
#pragma unroll
        for (int off = 16; off >= 1; off >>= 1) {
            const float ov = __shfl_xor(bestv, off);
            const int   oi = __shfl_xor(besti, off);
            if (ov > bestv || (ov == bestv && oi < besti)) { bestv = ov; besti = oi; }
        }
        if (lane == 0) { rv[wid] = bestv; ri[wid] = besti; }
        __syncthreads();
        if (tid == 0) {
            float v = rv[0]; int i = ri[0];
#pragma unroll
            for (int w = 1; w < 8; ++w) {
                if (rv[w] > v || (rv[w] == v && ri[w] < i)) { v = rv[w]; i = ri[w]; }
            }
            lastIdx = i;
        }
        __syncthreads();
        cur = lastIdx;
    }
}

// ---------------------------------------------------------------------------
// Kernel 2: ball query. One wave32 per query; scan points in index order in
// chunks of 32, ballot + prefix-popcount keeps "first K_NEIGH within radius"
// ordering identical to the reference; zero-pad the tail.
// ---------------------------------------------------------------------------
__global__ __launch_bounds__(256) void ballq_kernel(const float* __restrict__ pts,
                                                    const float* __restrict__ selected,
                                                    float* __restrict__ grouped) {
    const int wid  = threadIdx.x >> 5;
    const int lane = threadIdx.x & 31;
    const int g    = blockIdx.x * 8 + wid;           // 8192 queries
    const int b    = g >> 10;

    const float* P = pts + (size_t)b * NB_N * 3;
    const float* Q = selected + (size_t)g * 3;
    const float qx = Q[0], qy = Q[1], qz = Q[2];
    float* og = grouped + (size_t)g * NB_K * 3;

    int cnt = 0;
    for (int base = 0; base < NB_N && cnt < NB_K; base += 32) {
        const int i = base + lane;
        const float x = P[i * 3 + 0], y = P[i * 3 + 1], z = P[i * 3 + 2];
        const float dx = x - qx, dy = y - qy, dz = z - qz;
        const float d2 = dx * dx + dy * dy + dz * dz;
        const bool valid = d2 < R2;
        const unsigned mask = __builtin_amdgcn_ballot_w32(valid);
        if (valid) {
            const int slot = cnt + __popc(mask & ((1u << lane) - 1u));
            if (slot < NB_K) {
                og[slot * 3 + 0] = x;
                og[slot * 3 + 1] = y;
                og[slot * 3 + 2] = z;
            }
        }
        cnt += __popc(mask);
    }
    if (cnt > NB_K) cnt = NB_K;
    for (int s = cnt + lane; s < NB_K; s += 32) {
        og[s * 3 + 0] = 0.f; og[s * 3 + 1] = 0.f; og[s * 3 + 2] = 0.f;
    }
}

// ---------------------------------------------------------------------------
// WMMA helpers (v_wmma_f32_16x16x32_f16 operand layouts per CDNA5 ISA 7.12.2)
// ---------------------------------------------------------------------------
// A fragment: 16x32 f16. VGPR v in 0..3: K = khalf*8 + 2(v&3); v in 4..7: +16.
// Each 4-VGPR half is 16 contiguous bytes -> two ds_load_b128 per fragment.
__device__ inline v16h load_a(const _Float16* act, int mbase, int kbase, int lane) {
    union { v16h h; unsigned u[8]; } A;
    const _Float16* p = act + (mbase + (lane & 15)) * 64;
    const int khalf = (lane >> 4) & 1;
#pragma unroll
    for (int v = 0; v < 8; ++v) {
        const int k = kbase + ((v & 4) ? 16 : 0) + khalf * 8 + 2 * (v & 3);
        A.u[v] = *(const unsigned*)(p + k);       // 4B-aligned pair of halves
    }
    return A.h;
}

// B fragment: 32x16 f16 (KxN). Lane&15 = N column, vector index = K row (same K order as A).
__device__ inline v16h load_b(const _Float16* w, int ldw, int kbase, int n0, int lane) {
    union { v16h h; unsigned u[8]; } B;
    const _Float16* p = w + kbase * ldw + n0 + (lane & 15);
    const int khalf = (lane >> 4) & 1;
#pragma unroll
    for (int s = 0; s < 16; ++s) B.h[s] = p[(((s & 8) ? 16 : 0) + khalf * 8 + (s & 7)) * ldw];
    return B.h;
}

// C/D tile: lane col = n0 + (lane&15); rows = mbase + (lane>=16 ? 8 : 0) + v.
__device__ inline void store_relu(_Float16* act, int mbase, int n0, int lane,
                                  v8f c, float bias) {
    const int col   = n0 + (lane & 15);
    const int rbase = mbase + ((lane >> 4) ? 8 : 0);
#pragma unroll
    for (int v = 0; v < 8; ++v) {
        const float x = c[v] + bias;
        act[(rbase + v) * 64 + col] = (_Float16)(x > 0.f ? x : 0.f);
    }
}

#define WMMA_F16(A, B, C) \
    __builtin_amdgcn_wmma_f32_16x16x32_f16(false, (A), false, (B), (short)0, (C), false, false)

// ---------------------------------------------------------------------------
// Kernel 3: shared MLP 3->64->64->128 (+bias,+ReLU) and max over 64 neighbors,
// fully on v_wmma_f32_16x16x32_f16. One wave per group (M=64 rows). Weights
// converted to f16 once per block into LDS; activations in a private 64x64
// half tile per wave (A-fragments fully in registers before overwrite).
// Per n-column: all 4 independent m-tile WMMAs are issued back-to-back into
// separate accumulators, epilogues run afterwards -> no dead v_nop hazard
// padding between WMMA and its consumer (WMMA->VALU RAW, ISA 7.12.1).
// ---------------------------------------------------------------------------
__global__ __launch_bounds__(128) void mlp_kernel(const float* __restrict__ grouped,
                                                  const float* __restrict__ w0,
                                                  const float* __restrict__ b0,
                                                  const float* __restrict__ w1,
                                                  const float* __restrict__ b1,
                                                  const float* __restrict__ w2,
                                                  const float* __restrict__ b2,
                                                  float* __restrict__ out) {
    __shared__ _Float16 sW0[32 * 64];       // 3x64 zero-padded to K=32
    __shared__ _Float16 sW1[64 * 64];
    __shared__ _Float16 sW2[64 * 128];
    __shared__ float    sB0[64], sB1[64], sB2[128];
    __shared__ _Float16 sAct[4][64 * 64];   // one 64x64 tile per wave

    const int tid = threadIdx.x;
    for (int i = tid; i < 32 * 64; i += 128) {
        const int k = i >> 6, n = i & 63;
        sW0[i] = (k < 3) ? (_Float16)w0[k * 64 + n] : (_Float16)0.f;
    }
    for (int i = tid; i < 64 * 64;  i += 128) sW1[i] = (_Float16)w1[i];
    for (int i = tid; i < 64 * 128; i += 128) sW2[i] = (_Float16)w2[i];
    for (int i = tid; i < 64;  i += 128) { sB0[i] = b0[i]; sB1[i] = b1[i]; }
    for (int i = tid; i < 128; i += 128) sB2[i] = b2[i];
    __syncthreads();

    const int wid  = tid >> 5;
    const int lane = tid & 31;
    const int g    = blockIdx.x * 4 + wid;           // group = b*1024 + q
    _Float16* act  = &sAct[wid][0];
    const float* gp = grouped + (size_t)g * NB_K * 3;
    __builtin_prefetch(gp, 0, 1);                    // global_prefetch_b8

    // Stage X: 64 rows x (3 real + 29 zero) halves -> cols 0..31
    for (int r = lane; r < NB_K; r += 32) {
        const float* pr = gp + r * 3;
        _Float16* row = act + r * 64;
        row[0] = (_Float16)pr[0];
        row[1] = (_Float16)pr[1];
        row[2] = (_Float16)pr[2];
#pragma unroll
        for (int c = 3; c < 32; ++c) row[c] = (_Float16)0.f;
    }

    // ---- layer 0: (64x32) x (32x64), 16 WMMAs -----------------------------
    {
        v16h A[4];
#pragma unroll
        for (int m = 0; m < 4; ++m) A[m] = load_a(act, m * 16, 0, lane);
#pragma unroll
        for (int n = 0; n < 4; ++n) {
            const v16h B = load_b(sW0, 64, 0, n * 16, lane);
            const float bias = sB0[n * 16 + (lane & 15)];
            v8f c[4];
#pragma unroll
            for (int m = 0; m < 4; ++m) {           // 4 independent WMMAs
                v8f z = {};
                c[m] = WMMA_F16(A[m], B, z);
            }
#pragma unroll
            for (int m = 0; m < 4; ++m)             // epilogues after pipeline
                store_relu(act, m * 16, n * 16, lane, c[m], bias);
        }
    }

    // ---- layer 1: (64x64) x (64x64), 32 WMMAs -----------------------------
    {
        v16h A[4][2];
#pragma unroll
        for (int m = 0; m < 4; ++m)
#pragma unroll
            for (int k = 0; k < 2; ++k) A[m][k] = load_a(act, m * 16, k * 32, lane);
#pragma unroll
        for (int n = 0; n < 4; ++n) {
            const v16h B0 = load_b(sW1, 64, 0,  n * 16, lane);
            const v16h B1 = load_b(sW1, 64, 32, n * 16, lane);
            const float bias = sB1[n * 16 + (lane & 15)];
            v8f c[4];
#pragma unroll
            for (int m = 0; m < 4; ++m) {           // 8 chained-pair WMMAs
                v8f z = {};
                z = WMMA_F16(A[m][0], B0, z);
                c[m] = WMMA_F16(A[m][1], B1, z);
            }
#pragma unroll
            for (int m = 0; m < 4; ++m)
                store_relu(act, m * 16, n * 16, lane, c[m], bias);
        }
    }

    // ---- layer 2: (64x64) x (64x128), 64 WMMAs, fused maxpool over M ------
    {
        v16h A[4][2];
#pragma unroll
        for (int m = 0; m < 4; ++m)
#pragma unroll
            for (int k = 0; k < 2; ++k) A[m][k] = load_a(act, m * 16, k * 32, lane);
        float* orow = out + (size_t)g * OUT_CH + 3;
#pragma unroll
        for (int n = 0; n < 8; ++n) {
            const v16h B0 = load_b(sW2, 128, 0,  n * 16, lane);
            const v16h B1 = load_b(sW2, 128, 32, n * 16, lane);
            const float bias = sB2[n * 16 + (lane & 15)];
            v8f c[4];
#pragma unroll
            for (int m = 0; m < 4; ++m) {           // 8 back-to-back WMMAs
                v8f z = {};
                z = WMMA_F16(A[m][0], B0, z);
                c[m] = WMMA_F16(A[m][1], B1, z);
            }
            float cmax = 0.f;                       // post-ReLU values are >= 0
#pragma unroll
            for (int m = 0; m < 4; ++m)
#pragma unroll
                for (int v = 0; v < 8; ++v) {
                    const float x = c[m][v] + bias;
                    cmax = fmaxf(cmax, x > 0.f ? x : 0.f);
                }
            cmax = fmaxf(cmax, __shfl_xor(cmax, 16));   // merge M halves (C/D layout)
            if (lane < 16) orow[n * 16 + lane] = cmax;
        }
    }
}

// ---------------------------------------------------------------------------
extern "C" void kernel_launch(void* const* d_in, const int* in_sizes, int n_in,
                              void* d_out, int out_size, void* d_ws, size_t ws_size,
                              hipStream_t stream) {
    (void)in_sizes; (void)n_in; (void)out_size; (void)ws_size;
    const float* points = (const float*)d_in[0];
    const float* w0 = (const float*)d_in[1];
    const float* b0 = (const float*)d_in[2];
    const float* w1 = (const float*)d_in[3];
    const float* b1 = (const float*)d_in[4];
    const float* w2 = (const float*)d_in[5];
    const float* b2 = (const float*)d_in[6];
    float* out = (float*)d_out;

    float* selected = (float*)d_ws;                                  //  98,304 B
    float* grouped  = (float*)((char*)d_ws + (size_t)NB_B * NB_S * 3 * sizeof(float));
                                                                     // 6,291,456 B
    fps_kernel  <<<NB_B, 256, 0, stream>>>(points, selected, out);
    ballq_kernel<<<(NB_B * NB_S) / 8, 256, 0, stream>>>(points, selected, grouped);
    mlp_kernel  <<<(NB_B * NB_S) / 4, 128, 0, stream>>>(grouped, w0, b0, w1, b1,
                                                        w2, b2, out);
}
